// FourierAttention_23862838296639
// MI455X (gfx1250) — matrix operand order repaired and checked by reading, not compiled, not run on hardware
//
#include <hip/hip_runtime.h>
#include <hip/hip_bf16.h>

// ---- problem constants ----
#define BB   2
#define SS   2048
#define DD   1024
#define HH   16
#define DKK  64
#define INV_SCALE 0.125f   // 1/sqrt(64)

#define LDSK 48            // padded LDS row stride (bf16 elems): 96B rows, 16B aligned chunks

typedef __attribute__((ext_vector_type(16))) __bf16 bf16x16;
typedef __attribute__((ext_vector_type(8)))  __bf16 bf16x8;
typedef __attribute__((ext_vector_type(8)))  float  f32x8;

union AFrag { bf16x16 v; bf16x8 h[2]; };

// ---------------------------------------------------------------------------
// One K=32 WMMA step over a 128 x (NS*32) block. 8 waves = 4 (M) x 2 (N).
// ---------------------------------------------------------------------------
template <int NS>
__device__ __forceinline__ void mma_step(const __bf16* lA, const __bf16* lB,
                                         f32x8 (&acc)[2][NS],
                                         int wr, int wc, int hf, int ln) {
  AFrag a[2];
#pragma unroll
  for (int i = 0; i < 2; ++i) {
    int row = wr * 32 + i * 16 + ln;
    a[i].h[0] = *(const bf16x8*)(lA + row * LDSK + 8 * hf);        // K = 8*hf+e
    a[i].h[1] = *(const bf16x8*)(lA + row * LDSK + 16 + 8 * hf);   // K = 16+8*hf+e
  }
#pragma unroll
  for (int j = 0; j < NS; ++j) {
    int col = wc * (NS * 16) + j * 16 + ln;
    AFrag bf;
    bf.h[0] = *(const bf16x8*)(lB + col * LDSK + 8 * hf);          // K = 8*hf+e
    bf.h[1] = *(const bf16x8*)(lB + col * LDSK + 16 + 8 * hf);     // K = 16+8*hf+e
    bf16x16 b = bf.v;
#pragma unroll
    for (int i = 0; i < 2; ++i) {
      acc[i][j] = __builtin_amdgcn_wmma_f32_16x16x32_bf16(
          false, a[i].v, false, b, (short)0, acc[i][j], false, false);
    }
  }
}

// Async DMA one 128x32 bf16 tile (8KB) global -> LDS: 512 x 16B chunks, 2/thread.
__device__ __forceinline__ void fill128_async(__bf16* ldsbase, const __bf16* src,
                                              int ld, int k0, int tid) {
#pragma unroll
  for (int i = 0; i < 2; ++i) {
    int c = i * 256 + tid, row = c >> 2, part = c & 3;
    const __bf16* g = src + (size_t)row * ld + k0 + part * 8;
    unsigned lo = (unsigned)(uintptr_t)(ldsbase + row * LDSK + part * 8);
    asm volatile("global_load_async_to_lds_b128 %0, %1, off"
                 :: "v"(lo), "v"((unsigned long long)(uintptr_t)g) : "memory");
  }
}

// Async DMA one 64x32 bf16 tile: 256 x 16B chunks, 1/thread.
__device__ __forceinline__ void fill64_async(__bf16* ldsbase, const __bf16* src,
                                             int ld, int k0, int tid) {
  int row = tid >> 2, part = tid & 3;
  const __bf16* g = src + (size_t)row * ld + k0 + part * 8;
  unsigned lo = (unsigned)(uintptr_t)(ldsbase + row * LDSK + part * 8);
  asm volatile("global_load_async_to_lds_b128 %0, %1, off"
               :: "v"(lo), "v"((unsigned long long)(uintptr_t)g) : "memory");
}

// Synchronous stage of a 128x32 f32 tile into LDS as bf16 (attn operand only).
__device__ __forceinline__ void stage_cvt128(__bf16* lds, const float* src, int ld,
                                             int tid) {
#pragma unroll
  for (int i = 0; i < 4; ++i) {
    int e = (i * 256 + tid) * 4;
    int r = e >> 5, c = e & 31;
    const float4 v = *(const float4*)(src + (size_t)r * ld + c);
    __bf16* p = lds + r * LDSK + c;
    p[0] = (__bf16)v.x; p[1] = (__bf16)v.y; p[2] = (__bf16)v.z; p[3] = (__bf16)v.w;
  }
}

// ---------------------------------------------------------------------------
// Output index mapping.
// 0: plain [M x 1024]            1: [B,H,S,DK]        2: [B,H,DK,S]
// 3: per-z (b,h) -> O[b,s,h*DK+dk]
// ---------------------------------------------------------------------------
template <int MODE>
__device__ __forceinline__ size_t out_index(int m, int n, int z) {
  if constexpr (MODE == 0) {
    return (size_t)m * DD + n;
  } else if constexpr (MODE == 1) {
    int b = m >> 11, s = m & (SS - 1), h = n >> 6, dk = n & (DKK - 1);
    return (((size_t)(b * HH + h) * SS) + s) * DKK + dk;
  } else if constexpr (MODE == 2) {
    int b = m >> 11, s = m & (SS - 1), h = n >> 6, dk = n & (DKK - 1);
    return ((size_t)(b * HH + h) * DKK + dk) * SS + s;
  } else {
    int b = z >> 4, h = z & 15;
    return ((size_t)(b * SS + m)) * DD + h * DKK + n;
  }
}

// ---------------------------------------------------------------------------
// C = A @ B^T (+bias): bf16 A[M,K], B[N,K], both K-contiguous.
// 128x128 tile, async double-buffered LDS staging.
// ---------------------------------------------------------------------------
template <int MODE, typename OutT>
__global__ __launch_bounds__(256) void gemm_async(
    const __bf16* __restrict__ A, int lda, size_t zsA,
    const __bf16* __restrict__ Bm, int ldb, size_t zsB,
    const float* __restrict__ bias, OutT* __restrict__ C, int Kdim) {
  __shared__ __align__(16) __bf16 lA[2][128 * LDSK];
  __shared__ __align__(16) __bf16 lB[2][128 * LDSK];

  const int tid = threadIdx.x;
  const int m0 = blockIdx.y * 128, n0 = blockIdx.x * 128;
  const int z = blockIdx.z;
  A  += z * zsA + (size_t)m0 * lda;
  Bm += z * zsB + (size_t)n0 * ldb;

  const int wave = tid >> 5, lane = tid & 31, hf = lane >> 4, ln = lane & 15;
  const int wr = wave & 3, wc = wave >> 2;

  f32x8 acc[2][4] = {};

  const int nsteps = Kdim >> 5;
  fill128_async(lA[0], A, lda, 0, tid);
  fill128_async(lB[0], Bm, ldb, 0, tid);
  for (int s = 0; s < nsteps; ++s) {
    const int cur = s & 1;
    if (s + 1 < nsteps) {
      fill128_async(lA[cur ^ 1], A, lda, (s + 1) * 32, tid);
      fill128_async(lB[cur ^ 1], Bm, ldb, (s + 1) * 32, tid);
      // 4 fills just issued; async loads complete in order -> previous 4 done
      asm volatile("s_wait_asynccnt 0x4" ::: "memory");
    } else {
      asm volatile("s_wait_asynccnt 0x0" ::: "memory");
    }
    __syncthreads();
    mma_step<4>(lA[cur], lB[cur], acc, wr, wc, hf, ln);
    __syncthreads();  // all waves done reading before refill next iteration
  }

#pragma unroll
  for (int i = 0; i < 2; ++i)
#pragma unroll
    for (int j = 0; j < 4; ++j)
#pragma unroll
      for (int r = 0; r < 8; ++r) {
        int m = m0 + wr * 32 + i * 16 + hf * 8 + r;
        int n = n0 + wc * 64 + j * 16 + ln;
        float v = acc[i][j][r];
        if (bias) v += bias[n];
        C[out_index<MODE>(m, n, z)] = (OutT)v;
      }
}

// ---------------------------------------------------------------------------
// Combined scores, one (b,h) per blockIdx.z.  All operands pre-scaled bf16:
//   attn = Qs@Kb^T + Qr@Kr^T + Qi@Ki^T   (weights/scale folded into Qs,Qr,Qi)
// ---------------------------------------------------------------------------
__global__ __launch_bounds__(256) void scores_kernel(
    const __bf16* __restrict__ Qs, const __bf16* __restrict__ Kb,
    const __bf16* __restrict__ Qr, const __bf16* __restrict__ Qi,
    const __bf16* __restrict__ Kr, const __bf16* __restrict__ Ki,
    float* __restrict__ attn) {
  __shared__ __align__(16) __bf16 lA[2][128 * LDSK];
  __shared__ __align__(16) __bf16 lB[2][128 * LDSK];

  const int tid = threadIdx.x;
  const int z = blockIdx.z;
  const size_t hb = (size_t)z * SS * DKK;
  const int m0 = blockIdx.y * 128, n0 = blockIdx.x * 128;
  const __bf16* As[3] = {Qs + hb + (size_t)m0 * DKK, Qr + hb + (size_t)m0 * DKK,
                         Qi + hb + (size_t)m0 * DKK};
  const __bf16* Bs[3] = {Kb + hb + (size_t)n0 * DKK, Kr + hb + (size_t)n0 * DKK,
                         Ki + hb + (size_t)n0 * DKK};

  const int wave = tid >> 5, lane = tid & 31, hf = lane >> 4, ln = lane & 15;
  const int wr = wave & 3, wc = wave >> 2;

  f32x8 acc[2][4] = {};

  // 6 steps: (src 0..2) x (k0 in {0,32}), double-buffered
  fill128_async(lA[0], As[0], DKK, 0, tid);
  fill128_async(lB[0], Bs[0], DKK, 0, tid);
  for (int s = 0; s < 6; ++s) {
    const int cur = s & 1;
    if (s + 1 < 6) {
      int ns = s + 1, nsrc = ns >> 1, nk = (ns & 1) * 32;
      fill128_async(lA[cur ^ 1], As[nsrc], DKK, nk, tid);
      fill128_async(lB[cur ^ 1], Bs[nsrc], DKK, nk, tid);
      asm volatile("s_wait_asynccnt 0x4" ::: "memory");
    } else {
      asm volatile("s_wait_asynccnt 0x0" ::: "memory");
    }
    __syncthreads();
    mma_step<4>(lA[cur], lB[cur], acc, wr, wc, hf, ln);
    __syncthreads();
  }

  float* Cb = attn + (size_t)z * SS * SS;
#pragma unroll
  for (int i = 0; i < 2; ++i)
#pragma unroll
    for (int j = 0; j < 4; ++j)
#pragma unroll
      for (int r = 0; r < 8; ++r) {
        int m = m0 + wr * 32 + i * 16 + hf * 8 + r;
        int n = n0 + wc * 64 + j * 16 + ln;
        Cb[(size_t)m * SS + n] = acc[i][j][r];
      }
}

// ---------------------------------------------------------------------------
// O[b,s,h*DK+dk] = attn @ V.  A = attn (f32, staged w/ convert), B = Vt (bf16,
// async).  128x64 tile per block, blockIdx.z = b*H+h.  Output bf16.
// ---------------------------------------------------------------------------
__global__ __launch_bounds__(256) void attnv_kernel(
    const float* __restrict__ attn, const __bf16* __restrict__ Vt,
    __bf16* __restrict__ O) {
  __shared__ __align__(16) __bf16 lA[128 * LDSK];
  __shared__ __align__(16) __bf16 lB[64 * LDSK];

  const int tid = threadIdx.x;
  const int z = blockIdx.z;
  const int m0 = blockIdx.y * 128;
  const float* Ap = attn + (size_t)z * SS * SS + (size_t)m0 * SS;
  const __bf16* Bp = Vt + (size_t)z * DKK * SS;

  const int wave = tid >> 5, lane = tid & 31, hf = lane >> 4, ln = lane & 15;
  const int wr = wave & 3, wc = wave >> 2;

  f32x8 acc[2][2] = {};

  for (int k0 = 0; k0 < SS; k0 += 32) {
    __syncthreads();
    fill64_async(lB, Bp, SS, k0, tid);           // V tile via async DMA
    stage_cvt128(lA, Ap + k0, SS, tid);          // attn tile f32 -> bf16
    if (k0 + 32 < SS)
      __builtin_prefetch((const void*)(Ap + (size_t)(tid >> 1) * SS + k0 + 32), 0, 1);
    asm volatile("s_wait_asynccnt 0x0" ::: "memory");
    __syncthreads();
    mma_step<2>(lA, lB, acc, wr, wc, hf, ln);
  }

#pragma unroll
  for (int i = 0; i < 2; ++i)
#pragma unroll
    for (int j = 0; j < 2; ++j)
#pragma unroll
      for (int r = 0; r < 8; ++r) {
        int m = m0 + wr * 32 + i * 16 + hf * 8 + r;
        int n = wc * 32 + j * 16 + ln;
        O[out_index<3>(m, n, z)] = (__bf16)acc[i][j][r];
      }
}

// ---------------------------------------------------------------------------
// Elementwise f32 -> bf16 with optional device-scalar scale.
// ---------------------------------------------------------------------------
__global__ __launch_bounds__(256) void convert_kernel(const float* __restrict__ in,
                                                      __bf16* __restrict__ out, int n,
                                                      const float* __restrict__ sp,
                                                      float smul) {
  const float s = (sp ? sp[0] : 1.0f) * smul;
  int i = (blockIdx.x * 256 + threadIdx.x) * 4;
  if (i < n) {
    float4 v = *(const float4*)(in + i);
    out[i + 0] = (__bf16)(v.x * s);
    out[i + 1] = (__bf16)(v.y * s);
    out[i + 2] = (__bf16)(v.z * s);
    out[i + 3] = (__bf16)(v.w * s);
  }
}

// ---------------------------------------------------------------------------
// Length-2048 forward FFT (DIT radix-2 in LDS) of one strided channel, bf16 out
// with device-scalar scale.  blockIdx.x = z*DK + dk.
// ---------------------------------------------------------------------------
__global__ __launch_bounds__(256) void fft2048(const float* __restrict__ in,
                                               __bf16* __restrict__ outr,
                                               __bf16* __restrict__ outi,
                                               const float* __restrict__ sp,
                                               float smul) {
  __shared__ float re[SS];
  __shared__ float im[SS];
  const float scale = (sp ? sp[0] : 1.0f) * smul;
  const int ch = blockIdx.x;
  const int z = ch >> 6, dk = ch & (DKK - 1);
  const size_t base = (size_t)z * SS * DKK + dk;
  const int t = threadIdx.x;

  for (int i = t; i < SS; i += 256) {
    unsigned rev = __brev((unsigned)i) >> 21;  // 11-bit reverse
    re[rev] = in[base + (size_t)i * DKK];
    im[rev] = 0.0f;
  }
  __syncthreads();

  for (int len = 2; len <= SS; len <<= 1) {
    const int hlen = len >> 1;
    const float ang0 = -6.283185307179586f / (float)len;
    for (int j = t; j < SS / 2; j += 256) {
      int pos = j & (hlen - 1);
      int i0 = ((j ^ pos) << 1) | pos;
      int i1 = i0 + hlen;
      float s, c;
      __sincosf(ang0 * (float)pos, &s, &c);
      float xr = re[i1], xi = im[i1];
      float tr = c * xr - s * xi;
      float ti = c * xi + s * xr;
      re[i1] = re[i0] - tr; im[i1] = im[i0] - ti;
      re[i0] += tr;         im[i0] += ti;
    }
    __syncthreads();
  }

  for (int i = t; i < SS; i += 256) {
    outr[base + (size_t)i * DKK] = (__bf16)(re[i] * scale);
    outi[base + (size_t)i * DKK] = (__bf16)(im[i] * scale);
  }
}

// ---------------------------------------------------------------------------
// In-place row softmax over 2048 columns; one block per row.
// ---------------------------------------------------------------------------
__global__ __launch_bounds__(256) void softmax_kernel(float* __restrict__ attn) {
  __shared__ float red[256];
  float* p = attn + (size_t)blockIdx.x * SS;
  const int t = threadIdx.x;

  float v[8], mx = -3.4e38f;
#pragma unroll
  for (int i = 0; i < 8; ++i) { v[i] = p[t + i * 256]; mx = fmaxf(mx, v[i]); }
  red[t] = mx; __syncthreads();
  for (int s = 128; s > 0; s >>= 1) { if (t < s) red[t] = fmaxf(red[t], red[t + s]); __syncthreads(); }
  mx = red[0]; __syncthreads();

  float sum = 0.0f;
#pragma unroll
  for (int i = 0; i < 8; ++i) { v[i] = __expf(v[i] - mx); sum += v[i]; }
  red[t] = sum; __syncthreads();
  for (int s = 128; s > 0; s >>= 1) { if (t < s) red[t] += red[t + s]; __syncthreads(); }
  const float inv = 1.0f / red[0];
#pragma unroll
  for (int i = 0; i < 8; ++i) p[t + i * 256] = v[i] * inv;
}

// ---------------------------------------------------------------------------
extern "C" void kernel_launch(void* const* d_in, const int* in_sizes, int n_in,
                              void* d_out, int out_size, void* d_ws, size_t ws_size,
                              hipStream_t stream) {
  const float* x  = (const float*)d_in[0];
  const float* bq = (const float*)d_in[2];
  const float* bk = (const float*)d_in[4];
  const float* bv = (const float*)d_in[6];
  const float* bo = (const float*)d_in[8];
  const float* tw = (const float*)d_in[9];
  const float* fw = (const float*)d_in[10];

  const size_t P4M = 4194304;  // B*S*D elements
  const size_t P1M = 1048576;  // D*D elements

  char* w = (char*)d_ws;
  float*  Qf  = (float*)w;  w += P4M * 4;   // Q f32 [B,H,S,DK] (FFT input)
  float*  Kf  = (float*)w;  w += P4M * 4;   // K f32 [B,H,S,DK]
  __bf16* xb  = (__bf16*)w; w += P4M * 2;
  __bf16* Wqb = (__bf16*)w; w += P1M * 2;
  __bf16* Wkb = (__bf16*)w; w += P1M * 2;
  __bf16* Wvb = (__bf16*)w; w += P1M * 2;
  __bf16* Wob = (__bf16*)w; w += P1M * 2;
  __bf16* Qsb = (__bf16*)w; w += P4M * 2;   // Q * tw/8
  __bf16* Kb  = (__bf16*)w; w += P4M * 2;
  __bf16* Vtb = (__bf16*)w; w += P4M * 2;   // [B,H,DK,S]
  __bf16* Qrb = (__bf16*)w; w += P4M * 2;   // Re(FFT Q) * fw/8
  __bf16* Qib = (__bf16*)w; w += P4M * 2;
  __bf16* Krb = (__bf16*)w; w += P4M * 2;
  __bf16* Kib = (__bf16*)w; w += P4M * 2;
  __bf16* Ob  = (__bf16*)w; w += P4M * 2;   // attn@V head-concat, bf16

  float* out  = (float*)d_out;
  float* attn = out + P4M;

  dim3 blk(256);

  // one-time f32 -> bf16 conversions of GEMM operands
  convert_kernel<<<dim3(P4M / 1024), blk, 0, stream>>>(x, xb, (int)P4M, nullptr, 1.0f);
  convert_kernel<<<dim3(P1M / 1024), blk, 0, stream>>>((const float*)d_in[1], Wqb, (int)P1M, nullptr, 1.0f);
  convert_kernel<<<dim3(P1M / 1024), blk, 0, stream>>>((const float*)d_in[3], Wkb, (int)P1M, nullptr, 1.0f);
  convert_kernel<<<dim3(P1M / 1024), blk, 0, stream>>>((const float*)d_in[5], Wvb, (int)P1M, nullptr, 1.0f);
  convert_kernel<<<dim3(P1M / 1024), blk, 0, stream>>>((const float*)d_in[7], Wob, (int)P1M, nullptr, 1.0f);

  // projections (x @ W^T + b)
  gemm_async<1, float ><<<dim3(8, 32, 1), blk, 0, stream>>>(xb, DD, 0, Wqb, DD, 0, bq, Qf, DD);
  gemm_async<1, float ><<<dim3(8, 32, 1), blk, 0, stream>>>(xb, DD, 0, Wkb, DD, 0, bk, Kf, DD);
  gemm_async<2, __bf16><<<dim3(8, 32, 1), blk, 0, stream>>>(xb, DD, 0, Wvb, DD, 0, bv, Vtb, DD);

  // pre-scaled bf16 score operands
  convert_kernel<<<dim3(P4M / 1024), blk, 0, stream>>>(Qf, Qsb, (int)P4M, tw, INV_SCALE);
  convert_kernel<<<dim3(P4M / 1024), blk, 0, stream>>>(Kf, Kb, (int)P4M, nullptr, 1.0f);

  // FFT along seq per (b,h,dk) channel, bf16 out (fw/8 folded into Q side)
  fft2048<<<dim3(BB * HH * DKK), blk, 0, stream>>>(Qf, Qrb, Qib, fw, INV_SCALE);
  fft2048<<<dim3(BB * HH * DKK), blk, 0, stream>>>(Kf, Krb, Kib, nullptr, 1.0f);

  // combined scores -> attn (f32 output region)
  scores_kernel<<<dim3(16, 16, BB * HH), blk, 0, stream>>>(Qsb, Kb, Qrb, Qib, Krb, Kib, attn);

  // softmax rows in place
  softmax_kernel<<<dim3(BB * HH * SS), blk, 0, stream>>>(attn);

  // O = attn @ V   (bf16 out)
  attnv_kernel<<<dim3(1, 16, BB * HH), blk, 0, stream>>>(attn, Vtb, Ob);

  // out = O @ Wo^T + bo
  gemm_async<0, float><<<dim3(8, 32, 1), blk, 0, stream>>>(Ob, DD, 0, Wob, DD, 0, bo, out, DD);
}
